// Attention_31688268710508
// MI455X (gfx1250) — compile-verified, matrix operand order checked by reading, not run on hardware
//
#include <hip/hip_runtime.h>
#include <hip/hip_bf16.h>

// ---------------- types for WMMA fragments ----------------
typedef _Float16 v8h  __attribute__((ext_vector_type(8)));
typedef _Float16 v16h __attribute__((ext_vector_type(16)));
typedef float    v8f  __attribute__((ext_vector_type(8)));

union F16x16 { v16h v; v8h h[2]; };

// A/B fragment loader for v_wmma_f32_16x16x32_f16.
// Layout (ISA 7.12.2, 16-bit A 16x32): lane l holds row (l&15); halves 0..7 are
// K = k0 + 8*(l>=16) .. +7, halves 8..15 are K = k0+16+8*(l>=16) .. +7.
__device__ inline v16h ld_frag16(const _Float16* p) {
  F16x16 f;
  f.h[0] = *(const v8h*)(p);
  f.h[1] = *(const v8h*)(p + 16);
  return f.v;
}

__device__ inline v8f v8f_zero() {
  v8f z;
  #pragma unroll
  for (int i = 0; i < 8; ++i) z[i] = 0.0f;
  return z;
}

#define WMMA_F32_F16(A, B, C) \
  __builtin_amdgcn_wmma_f32_16x16x32_f16(false, (A), false, (B), (short)0, (C), false, false)

// ---------------- CDNA5 async global->LDS copy (ASYNCcnt path) ----------------
// Each lane copies 16B from its global address to its LDS offset.
// Generic pointers to __shared__ carry the LDS byte offset in addr[31:0]
// (ISA 10.2 aperture rules), so a truncating cast gives the DS offset.
__device__ inline void async_cp_b128(_Float16* lds_ptr, const _Float16* gptr) {
  unsigned int loff = (unsigned int)(size_t)lds_ptr;
  asm volatile("global_load_async_to_lds_b128 %0, %1, off"
               :: "v"(loff), "v"(gptr) : "memory");
}

#if __has_builtin(__builtin_amdgcn_s_wait_asynccnt)
#define WAIT_ASYNC(n) __builtin_amdgcn_s_wait_asynccnt(n)
#else
#define WAIT_ASYNC(n) asm volatile("s_wait_asynccnt %0" :: "i"(n) : "memory")
#endif

#if __has_builtin(__builtin_amdgcn_s_wait_tensorcnt)
#define WAIT_TENSOR(n) __builtin_amdgcn_s_wait_tensorcnt((short)(n))
#else
#define WAIT_TENSOR(n) asm volatile("s_wait_tensorcnt %0" :: "i"(n) : "memory")
#endif

// ---------------- CDNA5 Tensor Data Mover (TENSORcnt path) ----------------
// One descriptor-driven DMA moves a 64x32 f16 tile (row stride = K elems in
// global) into a packed row-major LDS tile. D# layout per ISA 08 §8.3/8.4:
//   group0: count=1 | lds_addr | global_addr | type=2
//   group1: data_size=1(2B), tensor_dim0=2048, tensor_dim1=0xFFFF,
//           tile_dim0=32, tile_dim1=64, tensor_dim0_stride=K
// This toolchain exposes the 6-arg builtin form:
//   (uint32x4 g0, int32x8 g1, int32x4, int32x4, int32x8, i32 cpol)
#if __has_builtin(__builtin_amdgcn_tensor_load_to_lds)
#define HAVE_TDM 1
typedef unsigned int u32x4  __attribute__((ext_vector_type(4)));
typedef int          i32x4v __attribute__((ext_vector_type(4)));
typedef int          i32x8v __attribute__((ext_vector_type(8)));

__device__ inline void tdm_load_64x32_f16(_Float16* lds_ptr, const _Float16* gptr,
                                          int row_stride_elems) {
  unsigned long long ga = (unsigned long long)(size_t)gptr;
  u32x4 g0;
  g0[0] = 1u;                                   // count=1 (valid), user mode
  g0[1] = (unsigned int)(size_t)lds_ptr;        // lds_addr (bytes)
  g0[2] = (unsigned int)ga;                     // global_addr[31:0]
  g0[3] = ((unsigned int)(ga >> 32) & 0x01ffffffu) | (2u << 30); // addr[56:32] | type=2
  i32x8v g1;
  g1[0] = 0x00010000;                  // wg_mask=0, data_size=1 (2 bytes)
  g1[1] = (int)(2048u << 16);          // tensor_dim0[15:0]=2048 (bits 63:48)
  g1[2] = (int)(0xFFFFu << 16);        // tensor_dim0 hi=0, tensor_dim1 lo16=0xFFFF
  g1[3] = (int)(32u << 16);            // tensor_dim1 hi=0, tile_dim0=32
  g1[4] = 64;                          // tile_dim1=64, tile_dim2=0
  g1[5] = row_stride_elems;            // tensor_dim0_stride[31:0]
  g1[6] = 0;                           // stride hi, tensor_dim1_stride lo
  g1[7] = 0;
  i32x4v gz = {0, 0, 0, 0};
  i32x8v gz8 = {0, 0, 0, 0, 0, 0, 0, 0};
  __builtin_amdgcn_tensor_load_to_lds(g0, g1, gz, gz, gz8, 0);
}
#endif

// ---------------- problem constants ----------------
#define BATCH 2
#define SEQ   2048
#define DIM   2048
#define HEADS 16
#define HD    128
#define RANK  8
#define NH    (HEADS * HD)      // 2048
#define ROWS  (BATCH * SEQ)     // 4096
#define LORA_SCALE 2.0f

// ---------------- conversion kernels ----------------
__global__ void cvt_f32_to_f16(const float* __restrict__ in,
                               _Float16* __restrict__ out, int n) {
  int i = blockIdx.x * 256 + threadIdx.x;
  if (i < n) out[i] = (_Float16)in[i];
}

// LDS-tiled transpose+convert: w [K=2048][N=2048] f32 -> wT [N][K] f16.
__global__ __launch_bounds__(256) void transpose_cvt_f16(
    const float* __restrict__ w, _Float16* __restrict__ wT) {
  __shared__ float tile[32][33];
  int k0 = blockIdx.x * 32;
  int n0 = blockIdx.y * 32;
  int tx = threadIdx.x & 31;
  int ty = threadIdx.x >> 5;  // 0..7
  #pragma unroll
  for (int i = 0; i < 4; ++i) {
    int r = ty + i * 8;
    tile[r][tx] = w[(size_t)(k0 + r) * DIM + n0 + tx];
  }
  __syncthreads();
  #pragma unroll
  for (int i = 0; i < 4; ++i) {
    int r = ty + i * 8;
    wT[(size_t)(n0 + r) * DIM + k0 + tx] = (_Float16)tile[tx][r];
  }
}

// ---------------- rank-8 LoRA left factor: xa = x @ [head_a|q_a|k_a|v_a] ----------------
__global__ __launch_bounds__(256) void lora_xa_kernel(
    const float* __restrict__ x,  const float* __restrict__ ha,
    const float* __restrict__ qa, const float* __restrict__ ka,
    const float* __restrict__ va, float* __restrict__ xa) {
  int t   = threadIdx.x;
  int row = blockIdx.x * 8 + (t >> 5);
  int c   = t & 31;
  const float* A = (c < 8) ? ha : (c < 16) ? qa : (c < 24) ? ka : va;
  int j = c & 7;
  const float* xr = x + (size_t)row * DIM;
  float s = 0.0f;
  for (int k = 0; k < DIM; ++k) s += xr[k] * A[k * RANK + j];
  xa[row * 32 + c] = s;
}

// ---------------- WMMA GEMM, double-buffered LDS staging ----------------
// A-tile staged via per-lane GLOBAL_LOAD_ASYNC_TO_LDS_B128 (ASYNCcnt);
// B-tile staged via one TENSOR_LOAD_TO_LDS DMA from wave 0 (TENSORcnt),
// falling back to the async path if the TDM builtin is unavailable.
// flag 0/1: Q/K -> LoRA epilogue, store [B,H,S,HD] f16
// flag 2:   V   -> LoRA epilogue, store transposed [B,H,HD,S] f16
// flag 3:   out = O @ wo -> f32 row-major
__global__ __launch_bounds__(128) void gemm_wmma(
    const _Float16* __restrict__ A, const _Float16* __restrict__ BT,
    const float* __restrict__ xa, const float* __restrict__ head_b,
    const float* __restrict__ gb, _Float16* __restrict__ outh,
    float* __restrict__ out32, int flag) {
  const int K = DIM;
  const int KSTEPS = K / 32;                       // 64
  __shared__ __align__(16) _Float16 smA[2][64 * 32];
  __shared__ __align__(16) _Float16 smB[2][64 * 32];

  int t    = threadIdx.x;
  int lane = t & 31;
  int wave = t >> 5;
  int n0b  = blockIdx.x * 64;
  int m0b  = blockIdx.y * 64;
  int hi   = (lane >= 16) ? 8 : 0;
  int lm   = lane & 15;

  // per-thread copy mapping: 256 16B chunks per 64x32 tile; thread t copies
  // chunks t and t+128. chunk c: row = c>>2, half-offset = (c&3)*8.
  int c0row = t >> 2;              int c0k = (t & 3) * 8;
  int c1row = (t + 128) >> 2;      int c1k = ((t + 128) & 3) * 8;

  v8f acc[4];
  #pragma unroll
  for (int i = 0; i < 4; ++i) acc[i] = v8f_zero();

  // ---- staging helpers ----
  auto stageA = [&](int buf, int k0) {
    async_cp_b128(&smA[buf][c0row * 32 + c0k], A + (size_t)(m0b + c0row) * K + k0 + c0k);
    async_cp_b128(&smA[buf][c1row * 32 + c1k], A + (size_t)(m0b + c1row) * K + k0 + c1k);
  };
#ifdef HAVE_TDM
  auto stageB = [&](int buf, int k0) {
    if (wave == 0)
      tdm_load_64x32_f16(&smB[buf][0], BT + (size_t)n0b * K + k0, K);
  };
#define STAGE_ASYNCS 2
#else
  auto stageB = [&](int buf, int k0) {
    async_cp_b128(&smB[buf][c0row * 32 + c0k], BT + (size_t)(n0b + c0row) * K + k0 + c0k);
    async_cp_b128(&smB[buf][c1row * 32 + c1k], BT + (size_t)(n0b + c1row) * K + k0 + c1k);
  };
#define STAGE_ASYNCS 4
#endif

  // prologue: stage buffer 0
  stageA(0, 0);
  stageB(0, 0);

  #pragma unroll 1
  for (int ks = 0; ks < KSTEPS; ++ks) {
    int cur = ks & 1;
    if (ks + 1 < KSTEPS) {
      stageA(cur ^ 1, (ks + 1) * 32);
      stageB(cur ^ 1, (ks + 1) * 32);
      WAIT_ASYNC(STAGE_ASYNCS);     // this wave's copies into `cur` done
#ifdef HAVE_TDM
      if (wave == 0) WAIT_TENSOR(1); // oldest TDM (into `cur`) done
#endif
    } else {
      WAIT_ASYNC(0);
#ifdef HAVE_TDM
      if (wave == 0) WAIT_TENSOR(0);
#endif
    }
    __syncthreads();   // all staging into `cur` visible to all waves

    const _Float16* sa = &smA[cur][0];
    const _Float16* sb = &smB[cur][0];
    v16h af = ld_frag16(sa + (wave * 16 + lm) * 32 + hi);
    #pragma unroll
    for (int i = 0; i < 4; ++i) {
      v16h bf = ld_frag16(sb + (i * 16 + lm) * 32 + hi);
      acc[i] = WMMA_F32_F16(af, bf, acc[i]);
    }
    __syncthreads();   // done reading `cur` before it is restaged next iter
  }

  // -------- epilogue --------
  int m0 = m0b + wave * 16;
  int mg = (lane >> 4) * 8;  // C frag: VGPR j holds row mg + j
  #pragma unroll
  for (int i = 0; i < 4; ++i) {
    int n = n0b + i * 16 + lm;
    int h = n >> 7, d = n & 127;
    #pragma unroll
    for (int j = 0; j < 8; ++j) {
      int r = m0 + mg + j;
      float v = acc[i][j];
      if (flag < 3) {
        const float* xr = xa + r * 32;
        float lv = 0.0f;
        #pragma unroll
        for (int q = 0; q < 8; ++q) {
          lv += xr[q] * head_b[q * NH + n];            // shared head LoRA
          lv += xr[8 + flag * 8 + q] * gb[q * HD + d]; // per-proj LoRA (bcast over heads)
        }
        v += LORA_SCALE * lv;
        int bb = r >> 11, s = r & 2047;
        if (flag == 2) {
          outh[(((size_t)(bb * HEADS + h) * HD + d) * SEQ) + s] = (_Float16)v; // V: [B,H,HD,S]
        } else {
          outh[(((size_t)(bb * HEADS + h) * SEQ + s) * HD) + d] = (_Float16)v; // Q/K: [B,H,S,HD]
        }
      } else {
        out32[(size_t)r * DIM + n] = v;
      }
    }
  }
}

// ---------------- RoPE (elementwise on [B,H,S,HD] f16) ----------------
__global__ void rope_kernel(_Float16* __restrict__ T, const float* __restrict__ fc) {
  int idx = blockIdx.x * 256 + threadIdx.x;           // over B*H*S*(HD/2)
  if (idx >= BATCH * HEADS * SEQ * (HD / 2)) return;
  int d2 = idx & 63;
  int s  = (idx >> 6) & 2047;
  size_t off = (size_t)(idx >> 6) * HD + 2 * d2;
  float c  = fc[(s * 64 + d2) * 2 + 0];
  float si = fc[(s * 64 + d2) * 2 + 1];
  float tr = (float)T[off], ti = (float)T[off + 1];
  T[off]     = (_Float16)(tr * c - ti * si);
  T[off + 1] = (_Float16)(tr * si + ti * c);
}

// ---------------- causal flash attention ----------------
// Grid: (B*H) * (S/64) blocks, 128 threads (4 waves), wave owns 16 q rows.
__global__ __launch_bounds__(128) void attn_kernel(
    const _Float16* __restrict__ Qh, const _Float16* __restrict__ Kh,
    const _Float16* __restrict__ Vt, _Float16* __restrict__ Oh) {
  __shared__ __align__(16) _Float16 pbuf[4][16 * 32];
  int lane = threadIdx.x & 31;
  int wave = threadIdx.x >> 5;
  int bh = blockIdx.x >> 5;          // S/64 = 32 q-tiles per (b,h)
  int qt = blockIdx.x & 31;
  int q0 = qt * 64 + wave * 16;
  int hi = (lane >= 16) ? 8 : 0;
  int lm = lane & 15;
  int mg = (lane >> 4) * 8;

  const _Float16* qbase = Qh + (size_t)bh * SEQ * HD;
  const _Float16* kbase = Kh + (size_t)bh * SEQ * HD;
  const _Float16* vbase = Vt + (size_t)bh * HD * SEQ;

  v16h aq[4];
  #pragma unroll
  for (int kk = 0; kk < 4; ++kk)
    aq[kk] = ld_frag16(qbase + (size_t)(q0 + lm) * HD + kk * 32 + hi);

  v8f acc[8];
  #pragma unroll
  for (int f = 0; f < 8; ++f) acc[f] = v8f_zero();
  float mrow[8], lrow[8];
  #pragma unroll
  for (int j = 0; j < 8; ++j) { mrow[j] = -3.0e38f; lrow[j] = 0.0f; }

  const float scale = 0.08838834764831845f;  // 1/sqrt(128)
  int jend = q0 + 16;                        // causal: keys <= q0+15

  #pragma unroll 1
  for (int j0 = 0; j0 < jend; j0 += 32) {
    v8f sc[2];
    #pragma unroll
    for (int tt = 0; tt < 2; ++tt) {
      sc[tt] = v8f_zero();
      #pragma unroll
      for (int kk = 0; kk < 4; ++kk) {
        v16h bk = ld_frag16(kbase + (size_t)(j0 + tt * 16 + lm) * HD + kk * 32 + hi);
        sc[tt] = WMMA_F32_F16(aq[kk], bk, sc[tt]);
      }
    }
    #pragma unroll
    for (int j = 0; j < 8; ++j) {
      int qrow = q0 + mg + j;
      float s0 = sc[0][j] * scale;
      float s1 = sc[1][j] * scale;
      if (j0 + lm > qrow)      s0 = -3.0e38f;
      if (j0 + 16 + lm > qrow) s1 = -3.0e38f;
      float mx = fmaxf(s0, s1);
      mx = fmaxf(mx, __shfl_xor(mx, 1, 32));
      mx = fmaxf(mx, __shfl_xor(mx, 2, 32));
      mx = fmaxf(mx, __shfl_xor(mx, 4, 32));
      mx = fmaxf(mx, __shfl_xor(mx, 8, 32));
      float mnew = fmaxf(mrow[j], mx);
      float corr = __expf(mrow[j] - mnew);
      float p0 = __expf(s0 - mnew);
      float p1 = __expf(s1 - mnew);
      sc[0][j] = p0;
      sc[1][j] = p1;
      float ls = p0 + p1;
      ls += __shfl_xor(ls, 1, 32);
      ls += __shfl_xor(ls, 2, 32);
      ls += __shfl_xor(ls, 4, 32);
      ls += __shfl_xor(ls, 8, 32);
      lrow[j] = lrow[j] * corr + ls;
      mrow[j] = mnew;
      #pragma unroll
      for (int f = 0; f < 8; ++f) acc[f][j] *= corr;
    }
    // P (C-frag layout) -> LDS -> A-frag layout (wave-private tile)
    _Float16* pb = pbuf[wave];
    #pragma unroll
    for (int j = 0; j < 8; ++j) {
      pb[(mg + j) * 32 + lm]      = (_Float16)sc[0][j];
      pb[(mg + j) * 32 + 16 + lm] = (_Float16)sc[1][j];
    }
    v16h pa = ld_frag16(pb + lm * 32 + hi);
    #pragma unroll
    for (int f = 0; f < 8; ++f) {
      v16h bv = ld_frag16(vbase + (size_t)(f * 16 + lm) * SEQ + j0 + hi);
      acc[f] = WMMA_F32_F16(pa, bv, acc[f]);
    }
  }

  int bb = bh >> 4, h = bh & 15;
  size_t obase = (size_t)bb * SEQ * NH + (size_t)h * HD;
  #pragma unroll
  for (int f = 0; f < 8; ++f) {
    #pragma unroll
    for (int j = 0; j < 8; ++j) {
      int qrow = q0 + mg + j;
      float v = acc[f][j] / lrow[j];
      Oh[obase + (size_t)qrow * NH + f * 16 + lm] = (_Float16)v;
    }
  }
}

// ---------------- host-side launch ----------------
extern "C" void kernel_launch(void* const* d_in, const int* in_sizes, int n_in,
                              void* d_out, int out_size, void* d_ws, size_t ws_size,
                              hipStream_t stream) {
  const float* x  = (const float*)d_in[0];
  const float* fc = (const float*)d_in[1];
  const float* wq = (const float*)d_in[2];
  const float* wk = (const float*)d_in[3];
  const float* wv = (const float*)d_in[4];
  const float* ha = (const float*)d_in[5];
  const float* hb = (const float*)d_in[6];
  const float* qa = (const float*)d_in[7];
  const float* qb = (const float*)d_in[8];
  const float* ka = (const float*)d_in[9];
  const float* kb = (const float*)d_in[10];
  const float* va = (const float*)d_in[11];
  const float* vb = (const float*)d_in[12];
  const float* wo = (const float*)d_in[13];
  float* out = (float*)d_out;

  char* ws = (char*)d_ws;
  _Float16* XH  = (_Float16*)ws;  ws += (size_t)ROWS * DIM * 2;
  _Float16* WQT = (_Float16*)ws;  ws += (size_t)DIM * NH * 2;
  _Float16* WKT = (_Float16*)ws;  ws += (size_t)DIM * NH * 2;
  _Float16* WVT = (_Float16*)ws;  ws += (size_t)DIM * NH * 2;
  _Float16* WOT = (_Float16*)ws;  ws += (size_t)NH * DIM * 2;
  float*    XA  = (float*)ws;     ws += (size_t)ROWS * 32 * 4;
  _Float16* QH  = (_Float16*)ws;  ws += (size_t)BATCH * HEADS * SEQ * HD * 2;
  _Float16* KH  = (_Float16*)ws;  ws += (size_t)BATCH * HEADS * SEQ * HD * 2;
  _Float16* VT  = (_Float16*)ws;  ws += (size_t)BATCH * HEADS * HD * SEQ * 2;
  _Float16* OH  = (_Float16*)ws;  ws += (size_t)ROWS * NH * 2;

  {
    int n = ROWS * DIM;
    cvt_f32_to_f16<<<(n + 255) / 256, 256, 0, stream>>>(x, XH, n);
  }
  {
    dim3 tgrid(DIM / 32, NH / 32);
    transpose_cvt_f16<<<tgrid, 256, 0, stream>>>(wq, WQT);
    transpose_cvt_f16<<<tgrid, 256, 0, stream>>>(wk, WKT);
    transpose_cvt_f16<<<tgrid, 256, 0, stream>>>(wv, WVT);
    transpose_cvt_f16<<<tgrid, 256, 0, stream>>>(wo, WOT);
  }
  lora_xa_kernel<<<ROWS / 8, 256, 0, stream>>>(x, ha, qa, ka, va, XA);

  dim3 ggrid(NH / 64, ROWS / 64);
  gemm_wmma<<<ggrid, 128, 0, stream>>>(XH, WQT, XA, hb, qb, QH, nullptr, 0);
  gemm_wmma<<<ggrid, 128, 0, stream>>>(XH, WKT, XA, hb, kb, KH, nullptr, 1);
  gemm_wmma<<<ggrid, 128, 0, stream>>>(XH, WVT, XA, hb, vb, VT, nullptr, 2);

  {
    int n = BATCH * HEADS * SEQ * (HD / 2);
    rope_kernel<<<(n + 255) / 256, 256, 0, stream>>>(QH, fc);
    rope_kernel<<<(n + 255) / 256, 256, 0, stream>>>(KH, fc);
  }

  attn_kernel<<<BATCH * HEADS * (SEQ / 64), 128, 0, stream>>>(QH, KH, VT, OH);

  dim3 fgrid(DIM / 64, ROWS / 64);
  gemm_wmma<<<fgrid, 128, 0, stream>>>(OH, WOT, nullptr, nullptr, nullptr, nullptr, out, 3);
}
